// FFN_21998822490618
// MI455X (gfx1250) — matrix operand order repaired and checked by reading, not compiled, not run on hardware
//
#include <hip/hip_runtime.h>

// ---------------- problem constants ----------------
#define Bb   4
#define Cc   64
#define C4c  256
#define Dd   16
#define Hh   256
#define Ww   256
#define HW   65536
#define PP   8

typedef __bf16 bf16;
typedef __attribute__((ext_vector_type(16))) __bf16 v16bf;
typedef __attribute__((ext_vector_type(8)))  __bf16 v8bf;
typedef __attribute__((ext_vector_type(8)))  float  v8f;

#if defined(__HIP_DEVICE_COMPILE__) && __has_builtin(__builtin_amdgcn_tensor_load_to_lds)
#define USE_TDM 1
#else
#define USE_TDM 0
#endif

#if USE_TDM
typedef __attribute__((ext_vector_type(4))) unsigned int u32x4;
typedef __attribute__((ext_vector_type(8))) int          i32x8;
typedef __attribute__((ext_vector_type(4))) int          i32x4;

// Issue one TDM load of a 3D tile (x fastest, then y, then z) of bf16 data:
//   global addr = ga + 2*(x + y*stride_row + z*stride_z), LDS dest contiguous.
// Descriptor packing per CDNA5 ISA 8.3/8.4 (D# group0/group1), group2 carries
// tensor_dim2. Tensor dims set huge so no OOB clipping (tiles are in-bounds).
// This toolchain uses the 6-arg builtin: (g0, g1, g2, g3, extra, cpol).
__device__ __forceinline__ void tdm_load_tile_bf16(
    unsigned lds_off, unsigned long long ga,
    int tile_x, int tile_y, int tile_z,
    long long stride_row, long long stride_z) {
  u32x4 g0 = { 1u,                                   // count=1, user mode
               lds_off,                               // lds_addr
               (unsigned)(ga & 0xFFFFFFFFull),        // global_addr[31:0]
               (unsigned)((ga >> 32) & 0x1FFFFFFu) | (2u << 30) }; // addr[56:32]|type=2
  const unsigned td0 = 0x40000000u, td1 = 0x40000000u;
  i32x8 g1;
  g1[0] = (1 << 16);                                  // data_size=1 (2 bytes)
  g1[1] = (int)((td0 & 0xFFFFu) << 16);               // abar=0 | tensor_dim0 lo16
  g1[2] = (int)(((td0 >> 16) & 0xFFFFu) | ((td1 & 0xFFFFu) << 16));
  g1[3] = (int)(((td1 >> 16) & 0xFFFFu) | ((unsigned)tile_x << 16));
  g1[4] = (int)(((unsigned)tile_y) | ((unsigned)tile_z << 16));
  g1[5] = (int)(stride_row & 0xFFFFFFFFll);
  g1[6] = (int)(((stride_row >> 32) & 0xFFFFll) | ((stride_z & 0xFFFFll) << 16));
  g1[7] = (int)((stride_z >> 16) & 0xFFFFFFFFll);
  i32x4 g2 = { 0x40000000, 0, 0, 0 };                 // tensor_dim2 huge
  i32x4 g3 = { 0, 0, 0, 0 };
  i32x8 gx = { 0, 0, 0, 0, 0, 0, 0, 0 };
  __builtin_amdgcn_tensor_load_to_lds(g0, g1, g2, g3, gx, 0);
}
#endif

__device__ __forceinline__ float gelu_exact(float x) {
  return 0.5f * x * (1.0f + erff(x * 0.70710678118654752f));
}

__device__ __forceinline__ v8f wmma_bf16(v16bf a, v16bf b, v8f c) {
  return __builtin_amdgcn_wmma_f32_16x16x32_bf16(false, a, false, b, (short)0, c,
                                                 false, false);
}

// A fragment (16x32 bf16), row-major [M][ld] source in LDS.
__device__ __forceinline__ v16bf load_a_frag(const bf16* base, int ld, int m0, int k0) {
  int lane = threadIdx.x & 31;
  int r = lane & 15, hi = lane >> 4;
  const bf16* p = base + (m0 + r) * ld + k0 + hi * 8;
  v8bf lo = *(const v8bf*)p;
  v8bf h8 = *(const v8bf*)(p + 16);
  v16bf f;
#pragma unroll
  for (int i = 0; i < 8; ++i) { f[i] = lo[i]; f[i + 8] = h8[i]; }
  return f;
}

// B fragment (32x16 bf16) from pixel-major [N][ld] source in LDS.
__device__ __forceinline__ v16bf load_b_frag(const bf16* base, int ld, int n0, int k0) {
  int lane = threadIdx.x & 31;
  int n = lane & 15, hi = lane >> 4;
  const bf16* p = base + (n0 + n) * ld + k0 + hi * 8;
  v8bf lo = *(const v8bf*)p;
  v8bf h8 = *(const v8bf*)(p + 16);
  v16bf f;
#pragma unroll
  for (int i = 0; i < 8; ++i) { f[i] = lo[i]; f[i + 8] = h8[i]; }
  return f;
}

// ---------------- kernel 1: LayerNorm2d -> bf16 xn ----------------
__global__ void ln_kernel(const float* __restrict__ x,
                          const float* __restrict__ lnw,
                          const float* __restrict__ lnb,
                          bf16* __restrict__ xnb) {
  int idx = blockIdx.x * 256 + threadIdx.x;
  int b = idx >> 16;
  int p = idx & (HW - 1);
  long base = (long)b * Cc * HW + p;
  float v[Cc];
  float s = 0.f, ss = 0.f;
#pragma unroll 8
  for (int c = 0; c < Cc; ++c) {
    float t = x[base + (long)c * HW];
    v[c] = t; s += t; ss += t * t;
  }
  float mu = s * (1.0f / Cc);
  float var = ss * (1.0f / Cc) - mu * mu;
  float rs = rsqrtf(var + 1e-6f);
#pragma unroll 8
  for (int c = 0; c < Cc; ++c) {
    float y = (v[c] - mu) * rs * lnw[c] + lnb[c];
    xnb[base + (long)c * HW] = (bf16)y;
  }
}

// ---------------- kernel 2: partial 3x3 conv (16ch) -> bf16 y1 ----------------
__global__ void pconv_kernel(const bf16* __restrict__ xnb,
                             const float* __restrict__ w,
                             const float* __restrict__ bias,
                             bf16* __restrict__ y1b) {
  int idx = blockIdx.x * 256 + threadIdx.x;
  int p = idx & (HW - 1);
  int co = (idx >> 16) & (Dd - 1);
  int b = idx >> 20;
  int y = p >> 8, xx = p & 255;
  float acc = bias[co];
#pragma unroll
  for (int ci = 0; ci < Dd; ++ci) {
    long base = ((long)(b * Cc + ci)) * HW;
#pragma unroll
    for (int dy = -1; dy <= 1; ++dy) {
      int yy = y + dy;
      if (yy < 0 || yy >= Hh) continue;
#pragma unroll
      for (int dx = -1; dx <= 1; ++dx) {
        int xc = xx + dx;
        if (xc < 0 || xc >= Ww) continue;
        acc += w[((co * Dd + ci) * 9) + (dy + 1) * 3 + (dx + 1)] *
               (float)xnb[base + yy * Ww + xc];
      }
    }
  }
  y1b[((long)(b * Dd + co)) * HW + p] = (bf16)acc;
}

// ---------------- kernel 3: fft weight -> circular-conv taps ----------------
__global__ void fftk_kernel(const float* __restrict__ fw, float* __restrict__ kout) {
  int c = blockIdx.x;
  int t = threadIdx.x;
  int dy = t >> 3, dx = t & 7;
  float acc = 0.f;
  for (int fy = 0; fy < PP; ++fy) {
    for (int fx = 0; fx < PP; ++fx) {
      float wv;
      if (fx <= PP / 2) wv = fw[(c * PP + fy) * (PP / 2 + 1) + fx];
      else              wv = fw[(c * PP + ((PP - fy) & 7)) * (PP / 2 + 1) + (PP - fx)];
      int ph = (fy * dy + fx * dx) & 7;
      acc += wv * cosf(0.78539816339744831f * (float)ph);
    }
  }
  kout[c * 64 + t] = acc * (1.0f / 64.0f);
}

// ---------------- kernel 4: branch1 fused MLP + windowed circular conv ----------------
__global__ __launch_bounds__(256) void branch1_kernel(
    const bf16* __restrict__ xnb, const float* __restrict__ w0,
    const float* __restrict__ b0, const float* __restrict__ w2,
    const float* __restrict__ b2, const float* __restrict__ k1,
    float* __restrict__ dout) {
  // phase0: sStage[64][256] @32768 (32K, TDM channel-major image)
  // phase1: sX[256][64] @0  sW0c @32768  sW2c @36864  sU @40960 (->56K)
  // phase2: sOut[64][256] @0 (64K);  sK @65536 (16K); total 80K
  __shared__ __align__(16) unsigned char smem[81920];
  bf16 (*sX)[Cc]     = (bf16(*)[Cc])(smem);
  bf16 (*sW0c)[Cc]   = (bf16(*)[Cc])(smem + 32768);
  bf16 (*sW2c)[32]   = (bf16(*)[32])(smem + 36864);
  bf16 (*sU)[32]     = (bf16(*)[32])(smem + 40960);
  float (*sOut)[256] = (float(*)[256])(smem);
  float* sK          = (float*)(smem + 65536);
  bf16* sStage       = (bf16*)(smem + 32768);

  int t = threadIdx.x;
  int wave = t >> 5;
  int lane = t & 31;
  int blk = blockIdx.x;
  int b = blk >> 8;
  int tile = blk & 255;
  int y0 = (tile >> 4) * 16, x0 = (tile & 15) * 16;

#pragma unroll
  for (int i = 0; i < 16; ++i) sK[t + i * 256] = k1[t + i * 256];

#if USE_TDM
  if (wave == 0) {
    unsigned lds_off = (unsigned)(unsigned long long)(size_t)(void*)sStage;
    unsigned long long ga =
        (unsigned long long)(size_t)(xnb + (size_t)b * Cc * HW + (size_t)y0 * Ww + x0);
    tdm_load_tile_bf16(lds_off, ga, 16, 16, Cc, Ww, HW);
    __builtin_amdgcn_s_wait_tensorcnt((short)0);
  }
  __syncthreads();
  // transpose channel-major stage -> pixel-major GEMM-B layout
  for (int c = 0; c < Cc; ++c) sX[t][c] = sStage[c * 256 + t];
#else
  {
    int py = t >> 4, px = t & 15;
    long gp = (long)b * Cc * HW + (y0 + py) * Ww + (x0 + px);
    for (int c = 0; c < Cc; ++c) sX[t][c] = xnb[gp + (long)c * HW];
  }
#endif

  v8f acc[8];
#pragma unroll
  for (int j = 0; j < 8; ++j) acc[j] = (v8f){};

  int mt = wave & 3;
  int ng = wave >> 2;

  for (int cc = 0; cc < 8; ++cc) {
    __syncthreads();
    if (cc + 1 < 8) {
      __builtin_prefetch(&w0[(cc + 1) * 32 * Cc], 0, 0);
      __builtin_prefetch(&w2[(cc + 1) * 32], 0, 0);
    }
#pragma unroll
    for (int i = 0; i < 8; ++i) {
      int idx = t + i * 256;
      int r0 = idx >> 6, k0 = idx & 63;
      sW0c[r0][k0] = (bf16)w0[(cc * 32 + r0) * Cc + k0];
      int r2 = idx >> 5, k2 = idx & 31;
      sW2c[r2][k2] = (bf16)w2[r2 * C4c + cc * 32 + k2];
    }
    __syncthreads();

    // expand: u[32][256] = GELU(W0c @ xn + b0c)
    {
      int mt2 = wave & 1;
      v16bf a0 = load_a_frag(&sW0c[0][0], Cc, mt2 * 16, 0);
      v16bf a1 = load_a_frag(&sW0c[0][0], Cc, mt2 * 16, 32);
#pragma unroll
      for (int j = 0; j < 4; ++j) {
        int nt = (wave >> 1) * 4 + j;
        v16bf bf0 = load_b_frag(&sX[0][0], Cc, nt * 16, 0);
        v16bf bf1 = load_b_frag(&sX[0][0], Cc, nt * 16, 32);
        v8f u = (v8f){};
        u = wmma_bf16(a0, bf0, u);
        u = wmma_bf16(a1, bf1, u);
        int n = lane & 15, mh = (lane >> 4) * 8;
        int px = nt * 16 + n;
#pragma unroll
        for (int r = 0; r < 8; ++r) {
          int chl = mt2 * 16 + mh + r;
          float v = u[r] + b0[cc * 32 + chl];
          sU[px][chl] = (bf16)gelu_exact(v);
        }
      }
    }
    __syncthreads();

    // contract: acc += W2c @ u
    {
      v16bf aw = load_a_frag(&sW2c[0][0], 32, mt * 16, 0);
#pragma unroll
      for (int j = 0; j < 8; ++j) {
        int nt = ng * 8 + j;
        v16bf bu = load_b_frag(&sU[0][0], 32, nt * 16, 0);
        acc[j] = wmma_bf16(aw, bu, acc[j]);
      }
    }
  }
  __syncthreads();

  {
    int n = lane & 15, mh = (lane >> 4) * 8;
#pragma unroll
    for (int j = 0; j < 8; ++j) {
      int px = (ng * 8 + j) * 16 + n;
#pragma unroll
      for (int r = 0; r < 8; ++r) {
        int ch = mt * 16 + mh + r;
        sOut[ch][px] = acc[j][r] + b2[ch];
      }
    }
  }
  __syncthreads();

  {
    int y = t >> 4, x = t & 15;
    int wy = (y >> 3) * 8, wx = (x >> 3) * 8;
    int iy = y & 7, ix = x & 7;
    long gp = (long)b * Cc * HW + (y0 + y) * Ww + (x0 + x);
    for (int c = 0; c < Cc; ++c) {
      float a = 0.f;
#pragma unroll
      for (int dy = 0; dy < 8; ++dy) {
        int sy = wy + ((iy + dy) & 7);
#pragma unroll
        for (int dx = 0; dx < 8; ++dx) {
          int sx = wx + ((ix + dx) & 7);
          a += sK[c * 64 + dy * 8 + dx] * sOut[c][sy * 16 + sx];
        }
      }
      dout[gp + (long)c * HW] = a;
    }
  }
}

// ---------------- kernel 5: branch2 fused (expand->GELU->dw3x3->GELU->contract)+conv ----------------
#define NH 324
#define NHP 336
__global__ __launch_bounds__(256) void branch2_kernel(
    const bf16* __restrict__ xnb, const bf16* __restrict__ y1b,
    const float* __restrict__ w1, const float* __restrict__ b1,
    const float* __restrict__ w3, const float* __restrict__ b3,
    const float* __restrict__ w4, const float* __restrict__ b4,
    const float* __restrict__ k2, float* __restrict__ dout) {
  // phase0: sStage2[64][324] @43008 (41472, TDM channel-major halo image)
  // phase1: sX2[336][64] @0  sT @43008  sV @64512  sW1c @80896  sW4c @84992  sW3c @89088
  // phase2: sOut[64][256] @0;  sK2 @90240;  total 106624
  __shared__ __align__(16) unsigned char smem[106624];
  bf16 (*sX2)[Cc]    = (bf16(*)[Cc])(smem);
  bf16 (*sT)[32]     = (bf16(*)[32])(smem + 43008);
  bf16 (*sV)[32]     = (bf16(*)[32])(smem + 64512);
  bf16 (*sW1c)[Cc]   = (bf16(*)[Cc])(smem + 80896);
  bf16 (*sW4c)[32]   = (bf16(*)[32])(smem + 84992);
  float* sW3c        = (float*)(smem + 89088);
  float (*sOut)[256] = (float(*)[256])(smem);
  float* sK          = (float*)(smem + 90240);
  bf16* sStage2      = (bf16*)(smem + 43008);

  int t = threadIdx.x;
  int wave = t >> 5;
  int lane = t & 31;
  int blk = blockIdx.x;
  int b = blk >> 8;
  int tile = blk & 255;
  int y0 = (tile >> 4) * 16, x0 = (tile & 15) * 16;

#pragma unroll
  for (int i = 0; i < 16; ++i) sK[t + i * 256] = k2[t + i * 256];

  bool interior = (y0 >= 1) && (y0 + 17 <= Hh) && (x0 >= 1) && (x0 + 17 <= Ww);
#if USE_TDM
  if (interior) {
    if (wave == 0) {
      unsigned lds0 = (unsigned)(unsigned long long)(size_t)(void*)sStage2;
      size_t sp = (size_t)(y0 - 1) * Ww + (x0 - 1);
      unsigned long long gy =
          (unsigned long long)(size_t)(y1b + (size_t)b * Dd * HW + sp);
      unsigned long long gx =
          (unsigned long long)(size_t)(xnb + ((size_t)b * Cc + Dd) * HW + sp);
      tdm_load_tile_bf16(lds0, gy, 18, 18, Dd, Ww, HW);                    // ch 0..15
      tdm_load_tile_bf16(lds0 + Dd * NH * 2, gx, 18, 18, Cc - Dd, Ww, HW); // ch 16..63
      __builtin_amdgcn_s_wait_tensorcnt((short)0);
    }
    __syncthreads();
#pragma unroll
    for (int i = 0; i < 2; ++i) {
      int hp = t + i * 256;
      if (hp < NH) {
        for (int c = 0; c < Cc; ++c) sX2[hp][c] = sStage2[c * NH + hp];
      } else if (hp < NHP) {
        for (int c = 0; c < Cc; ++c) sX2[hp][c] = (bf16)0.0f;
      }
    }
  } else
#endif
  {
    for (int i = 0; i < 2; ++i) {
      int hp = t + i * 256;
      if (hp >= NHP) break;
      if (hp < NH) {
        int hy = hp / 18, hx = hp % 18;
        int gy = y0 - 1 + hy, gx = x0 - 1 + hx;
        bool in = (gy >= 0) && (gy < Hh) && (gx >= 0) && (gx < Ww);
        for (int c = 0; c < Cc; ++c) {
          bf16 v = (bf16)0.0f;
          if (in) {
            if (c < Dd) v = y1b[((long)(b * Dd + c)) * HW + gy * Ww + gx];
            else        v = xnb[((long)(b * Cc + c)) * HW + gy * Ww + gx];
          }
          sX2[hp][c] = v;
        }
      } else {
        for (int c = 0; c < Cc; ++c) sX2[hp][c] = (bf16)0.0f;
      }
    }
  }

  v8f acc[8];
#pragma unroll
  for (int j = 0; j < 8; ++j) acc[j] = (v8f){};
  int mt = wave & 3;
  int ng = wave >> 2;

  for (int cc = 0; cc < 8; ++cc) {
    __syncthreads();
    if (cc + 1 < 8) {
      __builtin_prefetch(&w1[(cc + 1) * 32 * Cc], 0, 0);
      __builtin_prefetch(&w4[(cc + 1) * 32], 0, 0);
    }
#pragma unroll
    for (int i = 0; i < 8; ++i) {
      int idx = t + i * 256;
      int r0 = idx >> 6, k0 = idx & 63;
      sW1c[r0][k0] = (bf16)w1[(cc * 32 + r0) * Cc + k0];
      int r4 = idx >> 5, k4 = idx & 31;
      sW4c[r4][k4] = (bf16)w4[r4 * C4c + cc * 32 + k4];
    }
    if (t < 288) sW3c[t] = w3[cc * 32 * 9 + t];
    __syncthreads();

    // expand over halo: t2[336][32] = GELU(W1c @ x2cat + b1c)
    for (int idx = wave; idx < 42; idx += 8) {
      int mt2 = idx & 1;
      int nt = idx >> 1;
      v16bf a0 = load_a_frag(&sW1c[0][0], Cc, mt2 * 16, 0);
      v16bf a1 = load_a_frag(&sW1c[0][0], Cc, mt2 * 16, 32);
      v16bf bf0 = load_b_frag(&sX2[0][0], Cc, nt * 16, 0);
      v16bf bf1 = load_b_frag(&sX2[0][0], Cc, nt * 16, 32);
      v8f u = (v8f){};
      u = wmma_bf16(a0, bf0, u);
      u = wmma_bf16(a1, bf1, u);
      int n = lane & 15, mh = (lane >> 4) * 8;
      int px = nt * 16 + n;
#pragma unroll
      for (int r = 0; r < 8; ++r) {
        int chl = mt2 * 16 + mh + r;
        float v = u[r] + b1[cc * 32 + chl];
        sT[px][chl] = (bf16)gelu_exact(v);
      }
    }
    __syncthreads();

    // depthwise 3x3 + GELU on interior 16x16
    {
      int y = t >> 4, x = t & 15;
      int hy = y + 1, hx = x + 1;
#pragma unroll 4
      for (int ch = 0; ch < 32; ++ch) {
        float a = b3[cc * 32 + ch];
#pragma unroll
        for (int dy = 0; dy < 3; ++dy)
#pragma unroll
          for (int dx = 0; dx < 3; ++dx)
            a += sW3c[ch * 9 + dy * 3 + dx] *
                 (float)sT[(hy - 1 + dy) * 18 + (hx - 1 + dx)][ch];
        sV[t][ch] = (bf16)gelu_exact(a);
      }
    }
    __syncthreads();

    // contract: acc += W4c @ v
    {
      v16bf aw = load_a_frag(&sW4c[0][0], 32, mt * 16, 0);
#pragma unroll
      for (int j = 0; j < 8; ++j) {
        int nt = ng * 8 + j;
        v16bf bu = load_b_frag(&sV[0][0], 32, nt * 16, 0);
        acc[j] = wmma_bf16(aw, bu, acc[j]);
      }
    }
  }
  __syncthreads();

  {
    int n = lane & 15, mh = (lane >> 4) * 8;
#pragma unroll
    for (int j = 0; j < 8; ++j) {
      int px = (ng * 8 + j) * 16 + n;
#pragma unroll
      for (int r = 0; r < 8; ++r) {
        int ch = mt * 16 + mh + r;
        sOut[ch][px] = acc[j][r] + b4[ch];
      }
    }
  }
  __syncthreads();

  {
    int y = t >> 4, x = t & 15;
    int wy = (y >> 3) * 8, wx = (x >> 3) * 8;
    int iy = y & 7, ix = x & 7;
    long gp = (long)b * Cc * HW + (y0 + y) * Ww + (x0 + x);
    for (int c = 0; c < Cc; ++c) {
      float a = 0.f;
#pragma unroll
      for (int dy = 0; dy < 8; ++dy) {
        int sy = wy + ((iy + dy) & 7);
#pragma unroll
        for (int dx = 0; dx < 8; ++dx) {
          int sx = wx + ((ix + dx) & 7);
          a += sK[c * 64 + dy * 8 + dx] * sOut[c][sy * 16 + sx];
        }
      }
      long gi = gp + (long)c * HW;
      dout[gi] = dout[gi] + a;
    }
  }
}

// ---------------- launch ----------------
extern "C" void kernel_launch(void* const* d_in, const int* in_sizes, int n_in,
                              void* d_out, int out_size, void* d_ws, size_t ws_size,
                              hipStream_t stream) {
  const float* x     = (const float*)d_in[0];
  const float* ln_w  = (const float*)d_in[1];
  const float* ln_b  = (const float*)d_in[2];
  const float* pcw   = (const float*)d_in[3];
  const float* pcb   = (const float*)d_in[4];
  const float* w0    = (const float*)d_in[5];
  const float* b0    = (const float*)d_in[6];
  const float* w1    = (const float*)d_in[7];
  const float* b1    = (const float*)d_in[8];
  const float* w2    = (const float*)d_in[9];
  const float* b2    = (const float*)d_in[10];
  const float* w3    = (const float*)d_in[11];
  const float* b3    = (const float*)d_in[12];
  const float* w4    = (const float*)d_in[13];
  const float* b4    = (const float*)d_in[14];
  const float* fftw1 = (const float*)d_in[15];
  const float* fftw2 = (const float*)d_in[16];
  float* dout = (float*)d_out;

  char* ws = (char*)d_ws;
  bf16* xnb = (bf16*)ws;                                   // 32 MB
  bf16* y1b = (bf16*)(ws + 33554432);                      // 8 MB
  float* k1 = (float*)(ws + 33554432 + 8388608);           // 16 KB
  float* k2 = k1 + 4096;                                   // 16 KB

  ln_kernel<<<(Bb * HW) / 256, 256, 0, stream>>>(x, ln_w, ln_b, xnb);
  pconv_kernel<<<(Bb * Dd * HW) / 256, 256, 0, stream>>>(xnb, pcw, pcb, y1b);
  fftk_kernel<<<Cc, 64, 0, stream>>>(fftw1, k1);
  fftk_kernel<<<Cc, 64, 0, stream>>>(fftw2, k2);
  branch1_kernel<<<Bb * 256, 256, 0, stream>>>(xnb, w0, b0, w2, b2, k1, dout);
  branch2_kernel<<<Bb * 256, 256, 0, stream>>>(xnb, y1b, w1, b1, w3, b3, w4, b4,
                                               k2, dout);
}
